// Attention_36249523978322
// MI455X (gfx1250) — matrix operand order repaired and checked by reading, not compiled, not run on hardware
//
#include <hip/hip_runtime.h>
#include <math.h>

// ---------------------------------------------------------------------------
// Types for gfx1250 WMMA (wave32):  D(f32 16x16) = A(bf16 16x32) x B(bf16 32x16) + C
// ---------------------------------------------------------------------------
typedef __bf16 v16bf __attribute__((ext_vector_type(16)));
typedef float  v8f   __attribute__((ext_vector_type(8)));

union FragU {
    v16bf v;
    uint4 q[2];
    unsigned short s[16];
};

__device__ __forceinline__ unsigned short f2bf(float f) {
    unsigned int x = __float_as_uint(f);
    x += 0x7FFFu + ((x >> 16) & 1u);          // round-to-nearest-even
    return (unsigned short)(x >> 16);
}

__device__ __forceinline__ v8f zero8() {
    v8f z;
    #pragma unroll
    for (int i = 0; i < 8; ++i) z[i] = 0.0f;
    return z;
}

#define NEG_MAX (-3.402823466e38f)

// Problem constants
#define SQn   2048
#define Dn    2048
#define Hn    16
#define KVHn  4
#define DHn   128
#define NKn   2048
#define MEMn  1023
#define NKPAD 2080     // vT row: 15 zero guard + 2048 data + 17 zero guard
#define SCALEF 10.0f

// ---------------------------------------------------------------------------
// Generic GEMM: C[M,N] = A[M,K] @ B[N,K]^T   (A,B fp32 in global, staged to LDS
// as bf16; C fp32).  Block: 256 thr = 8 waves; C tile 128x128; wave = 32x64.
// ---------------------------------------------------------------------------
__global__ __launch_bounds__(256) void gemm_bf16_tn(
    float* __restrict__ C, const float* __restrict__ A, const float* __restrict__ B,
    int M, int N, int K)
{
    __shared__ unsigned short As[128 * 32];
    __shared__ unsigned short Bs[128 * 32];

    const int t    = threadIdx.x;
    const int lane = t & 31, wid = t >> 5;
    const int half = lane >> 4, rl = lane & 15;
    const int wm   = wid & 3;        // 4 waves along M (32 rows each)
    const int wn   = wid >> 2;       // 2 waves along N (64 cols each)
    const long m0  = (long)blockIdx.y * 128;
    const long n0  = (long)blockIdx.x * 128;

    v8f acc[2][4];
    #pragma unroll
    for (int i = 0; i < 2; ++i)
        #pragma unroll
        for (int j = 0; j < 4; ++j) acc[i][j] = zero8();

    const int srow = t >> 1, sh = t & 1;   // 2 threads per staged row, 16 bf16 each

    for (int k0 = 0; k0 < K; k0 += 32) {
        #pragma unroll
        for (int q = 0; q < 4; ++q) {
            const float4 fa = *(const float4*)(A + (m0 + srow) * K + k0 + sh * 16 + q * 4);
            ushort4 pa; pa.x = f2bf(fa.x); pa.y = f2bf(fa.y); pa.z = f2bf(fa.z); pa.w = f2bf(fa.w);
            *(ushort4*)&As[srow * 32 + sh * 16 + q * 4] = pa;
            const float4 fb = *(const float4*)(B + (n0 + srow) * K + k0 + sh * 16 + q * 4);
            ushort4 pb; pb.x = f2bf(fb.x); pb.y = f2bf(fb.y); pb.z = f2bf(fb.z); pb.w = f2bf(fb.w);
            *(ushort4*)&Bs[srow * 32 + sh * 16 + q * 4] = pb;
        }
        __syncthreads();

        FragU a[2], b[4];
        #pragma unroll
        for (int mi = 0; mi < 2; ++mi) {     // A 16x32: M=rl, K = 16*half + 0..15 contiguous
            const int off = (wm * 32 + mi * 16 + rl) * 32;
            a[mi].q[0] = *(const uint4*)&As[off + 16 * half];
            a[mi].q[1] = *(const uint4*)&As[off + 16 * half + 8];
        }
        #pragma unroll
        for (int ni = 0; ni < 4; ++ni) {     // B 32x16: N=rl, K = 16*half + 0..15 contiguous
            const int off = (wn * 64 + ni * 16 + rl) * 32 + 16 * half;
            b[ni].q[0] = *(const uint4*)&Bs[off];
            b[ni].q[1] = *(const uint4*)&Bs[off + 8];
        }
        #pragma unroll
        for (int mi = 0; mi < 2; ++mi)
            #pragma unroll
            for (int ni = 0; ni < 4; ++ni)
                acc[mi][ni] = __builtin_amdgcn_wmma_f32_16x16x32_bf16(
                    false, a[mi].v, false, b[ni].v, (short)0, acc[mi][ni], false, false);
        __syncthreads();
    }

    #pragma unroll
    for (int mi = 0; mi < 2; ++mi)
        #pragma unroll
        for (int ni = 0; ni < 4; ++ni)
            #pragma unroll
            for (int r = 0; r < 8; ++r) {
                const long row = m0 + wm * 32 + mi * 16 + r + 8 * half;
                const long col = n0 + wn * 64 + ni * 16 + rl;
                C[row * N + col] = acc[mi][ni][r];
            }
}

// ---------------------------------------------------------------------------
// Small elementwise / prep kernels
// ---------------------------------------------------------------------------
__global__ void zero_u16(unsigned short* p, long n) {
    const long i = (long)blockIdx.x * blockDim.x + threadIdx.x;
    if (i < n) p[i] = 0;
}

__global__ void concat_kv(const float* __restrict__ mem, const float* __restrict__ ctx,
                          float* __restrict__ kv) {
    const long idx = (long)blockIdx.x * blockDim.x + threadIdx.x;
    if (idx >= (long)NKn * Dn) return;
    const long row = idx >> 11, col = idx & 2047;
    kv[idx] = (row < MEMn) ? mem[row * Dn + col] : ctx[(row - MEMn) * Dn + col];
}

// l2norm -> per-(head,d) scale -> RoPE -> bf16, head-major output. One wave per row.
__global__ __launch_bounds__(256) void prep_qk(
    const float* __restrict__ pre, const float* __restrict__ scale,
    const float* __restrict__ freqs, unsigned short* __restrict__ outbf,
    int nH, int seqlen)
{
    const int t = threadIdx.x, lane = t & 31, wid = t >> 5;
    const int row = blockIdx.x * 8 + wid;
    if (row >= seqlen * nH) return;
    const int i = row / nH, h = row % nH;
    const int stride = nH * DHn;
    const int d0 = lane * 4;

    const float4 f = *(const float4*)(pre + (long)i * stride + h * DHn + d0);
    float ss = f.x * f.x + f.y * f.y + f.z * f.z + f.w * f.w;
    #pragma unroll
    for (int o = 16; o; o >>= 1) ss += __shfl_xor(ss, o, 32);
    const float inv = 1.0f / fmaxf(sqrtf(ss), 1e-12f);

    const float4 sc4 = *(const float4*)(scale + h * DHn + d0);
    const float v0 = f.x * inv * sc4.x, v1 = f.y * inv * sc4.y;
    const float v2 = f.z * inv * sc4.z, v3 = f.w * inv * sc4.w;

    const float2 fr = *(const float2*)(freqs + (long)i * (DHn / 2) + d0 / 2);
    const float c0 = cosf(fr.x), s0 = sinf(fr.x);
    const float c1 = cosf(fr.y), s1 = sinf(fr.y);
    ushort4 u;
    u.x = f2bf(v0 * c0 - v1 * s0);
    u.y = f2bf(v0 * s0 + v1 * c0);
    u.z = f2bf(v2 * c1 - v3 * s1);
    u.w = f2bf(v2 * s1 + v3 * c1);
    *(ushort4*)(outbf + ((long)h * seqlen + i) * DHn + d0) = u;
}

// v_pre [NK, KVH*DH] fp32 -> vT [KVH*DH][NKPAD] bf16 (guard-band padded)
__global__ void prep_v(const float* __restrict__ vpre, unsigned short* __restrict__ vTp) {
    const long idx = (long)blockIdx.x * blockDim.x + threadIdx.x;
    if (idx >= (long)NKn * (KVHn * DHn)) return;
    const long m = idx >> 9, c = idx & 511;
    vTp[c * NKPAD + 15 + m] = f2bf(vpre[idx]);
}

__global__ void gate_mul(const float* __restrict__ aout, const float* __restrict__ gpre,
                         const float* __restrict__ bg, float* __restrict__ y) {
    const long idx = (long)blockIdx.x * blockDim.x + threadIdx.x;
    if (idx >= (long)SQn * Dn) return;
    const int col = (int)(idx & 2047);
    const float gv = gpre[idx] + bg[col];
    y[idx] = aout[idx] * (1.0f / (1.0f + __expf(-gv)));
}

// ---------------------------------------------------------------------------
// Banded attention: one block per 16 queries. Keys: sink(j=0) + m in
// [base-15, base+16] (band).  QK & PV via WMMA; talking-heads / top-8 /
// softmax in LDS.  256 thr = 8 waves, wave w owns heads {2w, 2w+1}.
// ---------------------------------------------------------------------------
__global__ __launch_bounds__(256) void attn_kernel(
    const unsigned short* __restrict__ qbf,   // [H][SQ][DH]
    const unsigned short* __restrict__ kbf,   // [KVH][NK][DH]
    const unsigned short* __restrict__ vTp,   // [KVH*DH][NKPAD]
    const float* __restrict__ pre_talk, const float* __restrict__ post_talk,
    const float* __restrict__ head_scale,
    float* __restrict__ aout)                 // [SQ][H*DH]
{
    __shared__ float sc[Hn][16][36];                // scores, then attn weights
    __shared__ unsigned short atb[Hn][16 * 32];     // post-mixed attn, bf16

    const int t = threadIdx.x, lane = t & 31, wid = t >> 5;
    const int half = lane >> 4, rl = lane & 15;
    const int base = blockIdx.x * 16;

    // ---- QK: 16x32 banded score tile per head ----
    #pragma unroll
    for (int hh = 0; hh < 2; ++hh) {
        const int h = wid * 2 + hh;
        const int kvh = h >> 2;
        const unsigned short* qrow = qbf + ((long)h * SQn + base + rl) * DHn;
        #pragma unroll
        for (int c = 0; c < 2; ++c) {
            int m = base - 15 + c * 16 + rl;
            m = m < 0 ? 0 : (m > NKn - 1 ? NKn - 1 : m);   // clamp; masked later
            const unsigned short* krow = kbf + ((long)kvh * NKn + m) * DHn;
            v8f acc = zero8();
            #pragma unroll
            for (int ks = 0; ks < 4; ++ks) {
                FragU a, b;
                a.q[0] = *(const uint4*)(qrow + ks * 32 + 16 * half);
                a.q[1] = *(const uint4*)(qrow + ks * 32 + 16 * half + 8);
                b.q[0] = *(const uint4*)(krow + ks * 32 + 16 * half);
                b.q[1] = *(const uint4*)(krow + ks * 32 + 16 * half + 8);
                acc = __builtin_amdgcn_wmma_f32_16x16x32_bf16(
                    false, a.v, false, b.v, (short)0, acc, false, false);
            }
            #pragma unroll
            for (int r = 0; r < 8; ++r)
                sc[h][r + 8 * half][1 + c * 16 + rl] = acc[r] * SCALEF;
        }
    }
    __syncthreads();

    // ---- pre-talk mix + band mask + top-8 + softmax  (thread = one (g,row)) ----
    const int g = t & 15, r = t >> 4;
    float pcol[16];
    #pragma unroll
    for (int h = 0; h < 16; ++h) pcol[h] = pre_talk[h * 16 + g];

    float s[33];
    s[0] = 0.0f;                                   // sink: q . 0 = 0, mix of zeros = 0
    #pragma unroll
    for (int j = 1; j < 33; ++j) {
        float a = 0.0f;
        #pragma unroll
        for (int h = 0; h < 16; ++h) a += pcol[h] * sc[h][r][j];
        const int m = base - 16 + j;               // key index for column j
        const bool valid = (m >= 0) && (m < NKn) && (j >= r + 1) && (j <= r + 16);
        s[j] = valid ? a : NEG_MAX;
    }

    float tk[8];                                   // sorted top-8, descending
    #pragma unroll
    for (int p = 0; p < 8; ++p) tk[p] = NEG_MAX;
    #pragma unroll
    for (int j = 0; j < 33; ++j) {
        float v = s[j];
        #pragma unroll
        for (int p = 0; p < 8; ++p) {
            const float cur = tk[p];
            tk[p] = fmaxf(cur, v);
            v     = fminf(cur, v);
        }
    }
    const float thr = tk[7], mx = tk[0];

    float pw[33];
    float sum = 0.0f;
    #pragma unroll
    for (int j = 0; j < 33; ++j) {
        const float e = (s[j] < thr) ? 0.0f : __expf(s[j] - mx);
        pw[j] = e; sum += e;
    }
    const float rinv = 1.0f / sum;

    __syncthreads();                               // raw scores fully consumed
    #pragma unroll
    for (int j = 0; j < 33; ++j) sc[g][r][j] = pw[j] * rinv;
    __syncthreads();

    // ---- post-talk mix, convert to bf16 tiles for PV ----
    float qcol[16];
    #pragma unroll
    for (int h = 0; h < 16; ++h) qcol[h] = post_talk[h * 16 + g];
    #pragma unroll
    for (int j = 1; j < 33; ++j) {                 // j=0 (sink) has v=0: drop
        float a = 0.0f;
        #pragma unroll
        for (int h = 0; h < 16; ++h) a += qcol[h] * sc[h][r][j];
        atb[g][r * 32 + (j - 1)] = f2bf(a);
    }
    __syncthreads();

    // ---- PV: attn[16x32] @ v[32xDH] per head ----
    #pragma unroll
    for (int hh = 0; hh < 2; ++hh) {
        const int h = wid * 2 + hh;
        const int kvh = h >> 2;
        FragU a;
        const unsigned short* ar = &atb[h][rl * 32];
        a.q[0] = *(const uint4*)(ar + 16 * half);
        a.q[1] = *(const uint4*)(ar + 16 * half + 8);
        const float hs = head_scale[h];
        #pragma unroll
        for (int dt = 0; dt < 8; ++dt) {
            const unsigned short* vr =
                vTp + ((long)(kvh * DHn + dt * 16 + rl)) * NKPAD + base + 16 * half;
            FragU b;
            b.q[0] = *(const uint4*)vr;
            b.q[1] = *(const uint4*)(vr + 8);
            v8f acc = zero8();
            acc = __builtin_amdgcn_wmma_f32_16x16x32_bf16(
                false, a.v, false, b.v, (short)0, acc, false, false);
            #pragma unroll
            for (int rr = 0; rr < 8; ++rr)
                aout[(long)(base + rr + 8 * half) * Dn + h * DHn + dt * 16 + rl] = acc[rr] * hs;
        }
    }
}

// ---------------------------------------------------------------------------
// Host-side launch
// ---------------------------------------------------------------------------
extern "C" void kernel_launch(void* const* d_in, const int* in_sizes, int n_in,
                              void* d_out, int out_size, void* d_ws, size_t ws_size,
                              hipStream_t stream) {
    const float* x        = (const float*)d_in[0];
    const float* context  = (const float*)d_in[1];
    const float* mem      = (const float*)d_in[2];
    const float* freqs_q  = (const float*)d_in[3];
    const float* freqs_k  = (const float*)d_in[4];
    const float* Wq       = (const float*)d_in[5];
    const float* Wk       = (const float*)d_in[6];
    const float* Wv       = (const float*)d_in[7];
    const float* Wo       = (const float*)d_in[8];
    const float* Wg       = (const float*)d_in[9];
    const float* bg       = (const float*)d_in[10];
    const float* q_scale  = (const float*)d_in[11];
    const float* k_scale  = (const float*)d_in[12];
    const float* hscale   = (const float*)d_in[13];
    const float* pre_talk = (const float*)d_in[14];
    const float* post_talk= (const float*)d_in[15];
    (void)in_sizes; (void)n_in; (void)ws_size;

    char* ws = (char*)d_ws;
    const size_t A  = (size_t)SQn * Dn * sizeof(float);          // 16 MB
    float* kv_f32   = (float*)(ws + 0);                          // aliased w/ attn_out
    float* attn_out = (float*)(ws + 0);
    float* q_pre    = (float*)(ws + A);                          // aliased w/ y
    float* y        = (float*)(ws + A);
    float* g_pre    = (float*)(ws + 2 * A);
    float* k_pre    = (float*)(ws + 3 * A);
    float* v_pre    = (float*)(ws + 3 * A + (size_t)NKn * 512 * 4);
    unsigned short* q_bf = (unsigned short*)(ws + 3 * A + 2 * (size_t)NKn * 512 * 4);
    unsigned short* k_bf = (unsigned short*)(ws + 4 * A);
    unsigned short* vTp  = (unsigned short*)(ws + 4 * A + (size_t)KVHn * NKn * DHn * 2);

    // 1. zero vT guard bands, build kv = concat(mem, context)
    {
        const long nvt = (long)KVHn * DHn * NKPAD;
        zero_u16<<<(nvt + 255) / 256, 256, 0, stream>>>(vTp, nvt);
        concat_kv<<<((long)NKn * Dn + 255) / 256, 256, 0, stream>>>(mem, context, kv_f32);
    }
    // 2. projections (WMMA bf16 GEMMs)
    gemm_bf16_tn<<<dim3(Dn / 128, SQn / 128), 256, 0, stream>>>(q_pre, x, Wq, SQn, Dn, Dn);
    gemm_bf16_tn<<<dim3(Dn / 128, SQn / 128), 256, 0, stream>>>(g_pre, x, Wg, SQn, Dn, Dn);
    gemm_bf16_tn<<<dim3(512 / 128, NKn / 128), 256, 0, stream>>>(k_pre, kv_f32, Wk, NKn, 512, Dn);
    gemm_bf16_tn<<<dim3(512 / 128, NKn / 128), 256, 0, stream>>>(v_pre, kv_f32, Wv, NKn, 512, Dn);
    // 3. norm/scale/rope -> bf16 layouts
    prep_qk<<<(SQn * Hn) / 8, 256, 0, stream>>>(q_pre, q_scale, freqs_q, q_bf, Hn, SQn);
    prep_qk<<<(NKn * KVHn) / 8, 256, 0, stream>>>(k_pre, k_scale, freqs_k, k_bf, KVHn, NKn);
    prep_v<<<((long)NKn * 512 + 255) / 256, 256, 0, stream>>>(v_pre, vTp);
    // 4. banded talking-heads top-k attention
    attn_kernel<<<SQn / 16, 256, 0, stream>>>(q_bf, k_bf, vTp, pre_talk, post_talk,
                                              hscale, attn_out);
    // 5. gate, output projection
    gate_mul<<<((long)SQn * Dn + 255) / 256, 256, 0, stream>>>(attn_out, g_pre, bg, y);
    gemm_bf16_tn<<<dim3(Dn / 128, SQn / 128), 256, 0, stream>>>((float*)d_out, y, Wo,
                                                                SQn, Dn, Dn);
    (void)out_size;
}
